// Decoder_13649406066958
// MI455X (gfx1250) — matrix operand order repaired and checked by reading, not compiled, not run on hardware
//
#include <hip/hip_runtime.h>
#include <hip/hip_bf16.h>

// ---------------------------------------------------------------------------
// CDNA5 (gfx1250) transformer decoder layer.
// GEMMs + attention einsums: v_wmma_f32_16x16x32_bf16 (fp32 accum).
// Row-major tile staging uses GLOBAL_LOAD_ASYNC_TO_LDS_B128 (ASYNCcnt).
// ---------------------------------------------------------------------------

typedef __attribute__((ext_vector_type(16))) __bf16 v16bf;
typedef __attribute__((ext_vector_type(8)))  float  v8f;

union BFrag {                 // 16 bf16 = 8 dwords per lane
    v16bf        v;
    unsigned int u[8];
};

__device__ __forceinline__ v8f wmma_bf16(v16bf a, v16bf b, v8f c) {
    return __builtin_amdgcn_wmma_f32_16x16x32_bf16(
        /*neg_a=*/false, a, /*neg_b=*/false, b,
        /*c_mod=*/(short)0, c, /*reuse_a=*/false, /*reuse_b=*/false);
}

// Async global->LDS copy (CDNA5 Tensor/async path, tracked by ASYNCcnt).
// LDS operand: wave-relative byte offset = low 32 bits of the flat address.
__device__ __forceinline__ void async_load_b128(const void* gptr, void* lptr) {
    unsigned int       lds = (unsigned int)(unsigned long long)lptr;
    unsigned long long ga  = (unsigned long long)gptr;
    asm volatile("global_load_async_to_lds_b128 %0, %1, off"
                 :: "v"(lds), "v"(ga) : "memory");
}
__device__ __forceinline__ void wait_async() {
    asm volatile("s_wait_asynccnt 0x0" ::: "memory");
}

// float -> bf16 (RNE), raw u16 storage
__device__ __forceinline__ unsigned short f2bf(float f) {
    unsigned int u = __float_as_uint(f);
    if ((u & 0x7FFFFFFFu) > 0x7F800000u)            // NaN -> quiet NaN
        return (unsigned short)((u >> 16) | 0x40);
    return (unsigned short)((u + 0x7FFFu + ((u >> 16) & 1u)) >> 16);
}

// ---------------------------------------------------------------------------
// fp32 -> bf16 converter
// ---------------------------------------------------------------------------
__global__ void cvt_f32_bf16(const float* __restrict__ in,
                             unsigned short* __restrict__ out, long long n) {
    long long i = (long long)blockIdx.x * blockDim.x + threadIdx.x;
    if (i < n) out[i] = f2bf(in[i]);
}

// ---------------------------------------------------------------------------
// GEMM: C[M,N] = A[M,K](bf16) * W[K,N](bf16) + bias (fp32), optional ReLU.
// Workgroup: 256 threads (8 waves), tile 128(M) x 128(N), K-step 32.
// Waves 4(M) x 2(N); each wave owns 32x64 (2x4 WMMA tiles -> 8 WMMA/K-step).
// A tile staged via async global->LDS; B tile register-staged + transposed.
// ---------------------------------------------------------------------------
__global__ __launch_bounds__(256) void gemm_bf16_wmma(
    const unsigned short* __restrict__ A, const unsigned short* __restrict__ W,
    const float* __restrict__ bias,
    unsigned short* __restrict__ out_bf, float* __restrict__ out_f32,
    int M, int N, int K, int relu)
{
    __shared__ __attribute__((aligned(16))) unsigned short As[128][32];    // [m][k]
    __shared__ __attribute__((aligned(16))) unsigned short Bst[128][32];   // [n][k]

    const int tid   = threadIdx.x;
    const int lane  = tid & 31;
    const int wave  = tid >> 5;
    const int waveM = wave >> 1;          // 0..3 -> 32 rows each
    const int waveN = wave & 1;           // 0..1 -> 64 cols each
    const int half  = lane >> 4;          // 0/1
    const int lm    = lane & 15;
    const int row_base = blockIdx.y * 128;
    const int col_base = blockIdx.x * 128;

    v8f acc[2][4] = {};

    for (int kb = 0; kb < K; kb += 32) {
        // --- A tile (128x32): async global->LDS, 2 x b128 per thread ---
#pragma unroll
        for (int j = 0; j < 2; ++j) {
            const int i  = tid + 256 * j;      // 0..511
            const int r  = i >> 2;             // 0..127
            const int kp = (i & 3) * 8;        // 0,8,16,24
            async_load_b128(A + (size_t)(row_base + r) * K + kb + kp, &As[r][kp]);
        }
        // --- B tile (32x128): register-stage b128, transpose into [n][k] ---
        uint4 bt[2];
        int   bkr[2], bcp[2];
#pragma unroll
        for (int j = 0; j < 2; ++j) {
            const int i = tid + 256 * j;       // 0..511
            bkr[j] = i >> 4;                   // 0..31
            bcp[j] = (i & 15) * 8;             // 0..120
            bt[j]  = *(const uint4*)(W + (size_t)(kb + bkr[j]) * N + col_base + bcp[j]);
        }
#pragma unroll
        for (int j = 0; j < 2; ++j) {
            Bst[bcp[j] + 0][bkr[j]] = (unsigned short)(bt[j].x);
            Bst[bcp[j] + 1][bkr[j]] = (unsigned short)(bt[j].x >> 16);
            Bst[bcp[j] + 2][bkr[j]] = (unsigned short)(bt[j].y);
            Bst[bcp[j] + 3][bkr[j]] = (unsigned short)(bt[j].y >> 16);
            Bst[bcp[j] + 4][bkr[j]] = (unsigned short)(bt[j].z);
            Bst[bcp[j] + 5][bkr[j]] = (unsigned short)(bt[j].z >> 16);
            Bst[bcp[j] + 6][bkr[j]] = (unsigned short)(bt[j].w);
            Bst[bcp[j] + 7][bkr[j]] = (unsigned short)(bt[j].w >> 16);
        }
        if (kb + 32 < K) {   // hint next K-slab into L2 (global_prefetch_b8)
            __builtin_prefetch(A + (size_t)(row_base + (tid >> 1)) * K + kb + 32, 0, 1);
            __builtin_prefetch(W + (size_t)(kb + 32 + (tid & 31)) * N + col_base, 0, 1);
        }
        wait_async();
        __syncthreads();

        // --- A fragments (M=lane%16, K = half*16 + 2v, K pairs per VGPR) ---
        BFrag af[2];
#pragma unroll
        for (int mi = 0; mi < 2; ++mi) {
            const int r = waveM * 32 + mi * 16 + lm;
            const unsigned int* ap = (const unsigned int*)&As[r][0];
#pragma unroll
            for (int v = 0; v < 8; ++v) {
                const int pidx = half * 8 + v;
                af[mi].u[v] = ap[pidx];
            }
        }
        // --- B fragments (N=lane%16, K = half*16 + 2v), 8 WMMAs ---
#pragma unroll
        for (int ni = 0; ni < 4; ++ni) {
            const int c = waveN * 64 + ni * 16 + lm;
            const unsigned int* bp = (const unsigned int*)&Bst[c][0];
            BFrag bf;
#pragma unroll
            for (int v = 0; v < 8; ++v) bf.u[v] = bp[half * 8 + v];
            acc[0][ni] = wmma_bf16(af[0].v, bf.v, acc[0][ni]);
            acc[1][ni] = wmma_bf16(af[1].v, bf.v, acc[1][ni]);
        }
        __syncthreads();
    }

    // --- epilogue: +bias, optional ReLU, write bf16 and/or fp32 ---
#pragma unroll
    for (int ni = 0; ni < 4; ++ni) {
        const int c  = col_base + waveN * 64 + ni * 16 + lm;
        const float bv = bias ? bias[c] : 0.f;
#pragma unroll
        for (int mi = 0; mi < 2; ++mi) {
#pragma unroll
            for (int v = 0; v < 8; ++v) {
                const int r = row_base + waveM * 32 + mi * 16 + v + 8 * half;
                float val = acc[mi][ni][v] + bv;
                if (relu) val = fmaxf(val, 0.f);
                if (out_f32) out_f32[(size_t)r * N + c] = val;
                if (out_bf)  out_bf [(size_t)r * N + c] = f2bf(val);
            }
        }
    }
}

// ---------------------------------------------------------------------------
// Flash attention (per head, DH = 64). Q,K,V,O layout: [B*S, H*64] bf16.
// Workgroup: 256 threads = 8 waves, 128 query rows of one (b,h).
// Per 32-key block & wave: 4 WMMAs for Q.K^T, online softmax, 4 WMMAs for P.V.
// K tile staged async; V tile register-staged + transposed.
// ---------------------------------------------------------------------------
__global__ __launch_bounds__(256) void flash_attn_wmma(
    const unsigned short* __restrict__ Q, const unsigned short* __restrict__ Km,
    const unsigned short* __restrict__ Vm, unsigned short* __restrict__ O,
    int S, int D, int H, int causal)
{
    __shared__ __attribute__((aligned(16))) unsigned short Ks[32][64];   // [key][dh]
    __shared__ __attribute__((aligned(16))) unsigned short Vst[64][32];  // [dh][key]
    __shared__ float          sc[8][16][32];        // per-wave score scratch
    __shared__ unsigned short Ps[8][16][32];        // per-wave P (bf16)
    __shared__ float          alpha_s[8][16];
    __shared__ float          l_s[8][16];

    const int tid  = threadIdx.x;
    const int lane = tid & 31;
    const int wave = tid >> 5;
    const int half = lane >> 4;
    const int lm   = lane & 15;
    const int bh   = blockIdx.y;
    const int b    = bh / H;
    const int h    = bh % H;
    const int q0wg = blockIdx.x * 128;
    const int qt0  = q0wg + wave * 16;
    const size_t baseRow = (size_t)b * S;
    const int colQ = h * 64;
    const float scale = 0.125f;                     // 1/sqrt(64)

    // Q fragments for this wave's 16 rows: 2 K-chunks of 32 (DH=64), from global
    BFrag qf[2];
    {
        const int m = qt0 + lm;
        const unsigned int* qp = (const unsigned int*)(Q + (baseRow + m) * D + colQ);
#pragma unroll
        for (int c0 = 0; c0 < 2; ++c0) {
#pragma unroll
            for (int v = 0; v < 8; ++v) {
                const int kk = half * 16 + 2 * v;
                qf[c0].u[v] = qp[(c0 * 32 + kk) >> 1];
            }
        }
    }

    v8f  oacc[4] = {};
    float m_r = -__builtin_inff();   // lanes 0..15 own row lm
    float l_r = 0.f;

    const int kend = causal ? (q0wg + 128) : S;
    for (int kb = 0; kb < kend; kb += 32) {
        // --- K tile (32x64) async global->LDS: one b128 per thread ---
        {
            const int kr = tid >> 3;           // 0..31
            const int dp = (tid & 7) * 8;      // 0..56
            async_load_b128(Km + (baseRow + kb + kr) * D + colQ + dp, &Ks[kr][dp]);
            // --- V tile register-staged, transposed into [dh][key] ---
            const uint4 w = *(const uint4*)(Vm + (baseRow + kb + kr) * D + colQ + dp);
            Vst[dp + 0][kr] = (unsigned short)(w.x);
            Vst[dp + 1][kr] = (unsigned short)(w.x >> 16);
            Vst[dp + 2][kr] = (unsigned short)(w.y);
            Vst[dp + 3][kr] = (unsigned short)(w.y >> 16);
            Vst[dp + 4][kr] = (unsigned short)(w.z);
            Vst[dp + 5][kr] = (unsigned short)(w.z >> 16);
            Vst[dp + 6][kr] = (unsigned short)(w.w);
            Vst[dp + 7][kr] = (unsigned short)(w.w >> 16);
        }
        wait_async();
        __syncthreads();

        if (!causal || kb < qt0 + 16) {
            // ---- scores: S[16x32] = Q(16x64) . K^T(64x32) ----
            v8f s[2] = {};
#pragma unroll
            for (int t = 0; t < 2; ++t) {
                const int key = t * 16 + lm;
                const unsigned int* kpp = (const unsigned int*)&Ks[key][0];
#pragma unroll
                for (int c0 = 0; c0 < 2; ++c0) {
                    BFrag bf;
#pragma unroll
                    for (int v = 0; v < 8; ++v) bf.u[v] = kpp[c0 * 16 + half * 8 + v];
                    s[t] = wmma_bf16(qf[c0].v, bf.v, s[t]);
                }
            }
            // ---- masked, scaled scores -> LDS scratch ----
#pragma unroll
            for (int t = 0; t < 2; ++t) {
                const int keyg = kb + t * 16 + lm;
#pragma unroll
                for (int v = 0; v < 8; ++v) {
                    const int r = v + 8 * half;
                    float val = s[t][v] * scale;
                    if (causal && keyg > qt0 + r) val = -__builtin_inff();
                    sc[wave][r][t * 16 + lm] = val;
                }
            }
            // ---- online softmax: lanes 0..15 each own one row ----
            if (lane < 16) {
                float mx = m_r;
#pragma unroll
                for (int kk = 0; kk < 32; ++kk) mx = fmaxf(mx, sc[wave][lane][kk]);
                const float alpha = __expf(m_r - mx);
                float ssum = 0.f;
#pragma unroll
                for (int kk = 0; kk < 32; ++kk) {
                    const float p = __expf(sc[wave][lane][kk] - mx);
                    Ps[wave][lane][kk] = f2bf(p);
                    ssum += p;
                }
                m_r = mx;
                l_r = l_r * alpha + ssum;
                alpha_s[wave][lane] = alpha;
            }
            // ---- rescale accumulators (per-row alpha) ----
            float a8[8];
#pragma unroll
            for (int v = 0; v < 8; ++v) a8[v] = alpha_s[wave][v + 8 * half];
#pragma unroll
            for (int nt = 0; nt < 4; ++nt)
#pragma unroll
                for (int v = 0; v < 8; ++v) oacc[nt][v] *= a8[v];
            // ---- P fragment (A-layout from LDS) ----
            BFrag pf;
            {
                const unsigned int* pp = (const unsigned int*)&Ps[wave][lm][0];
#pragma unroll
                for (int v = 0; v < 8; ++v) {
                    const int pidx = half * 8 + v;
                    pf.u[v] = pp[pidx];
                }
            }
            // ---- O += P(16x32) . V(32x64) ----
#pragma unroll
            for (int nt = 0; nt < 4; ++nt) {
                const int dh = nt * 16 + lm;
                const unsigned int* vpp = (const unsigned int*)&Vst[dh][0];
                BFrag vf;
#pragma unroll
                for (int v = 0; v < 8; ++v) vf.u[v] = vpp[half * 8 + v];
                oacc[nt] = wmma_bf16(pf.v, vf.v, oacc[nt]);
            }
        }
        __syncthreads();
    }

    // ---- finalize: divide by row sums, write bf16 ----
    if (lane < 16) l_s[wave][lane] = l_r;
    float linv[8];
#pragma unroll
    for (int v = 0; v < 8; ++v) linv[v] = 1.0f / l_s[wave][v + 8 * half];
#pragma unroll
    for (int nt = 0; nt < 4; ++nt) {
        const int col = colQ + nt * 16 + lm;
#pragma unroll
        for (int v = 0; v < 8; ++v) {
            const int r = qt0 + v + 8 * half;
            O[(baseRow + r) * D + col] = f2bf(oacc[nt][v] * linv[v]);
        }
    }
}

// ---------------------------------------------------------------------------
// out = LayerNorm(x + res) * g + bt ; writes fp32 (residual chain) and
// optional bf16 (next GEMM input). One workgroup per row, D = 1024.
// ---------------------------------------------------------------------------
__global__ __launch_bounds__(256) void add_ln_kernel(
    const float* __restrict__ x, const float* __restrict__ res,
    const float* __restrict__ g, const float* __restrict__ bt,
    float* __restrict__ out_f32, unsigned short* __restrict__ out_bf, int D)
{
    __shared__ float red[256];
    const int m = blockIdx.x;
    const float* xr = x   + (size_t)m * D;
    const float* rr = res + (size_t)m * D;

    float vals[4];
    float s = 0.f, s2 = 0.f;
#pragma unroll
    for (int j = 0; j < 4; ++j) {
        const int i = threadIdx.x + 256 * j;
        const float v = xr[i] + rr[i];
        vals[j] = v; s += v; s2 += v * v;
    }
    red[threadIdx.x] = s;  __syncthreads();
    for (int st = 128; st > 0; st >>= 1) {
        if (threadIdx.x < st) red[threadIdx.x] += red[threadIdx.x + st];
        __syncthreads();
    }
    const float mu = red[0] / (float)D;
    __syncthreads();
    red[threadIdx.x] = s2; __syncthreads();
    for (int st = 128; st > 0; st >>= 1) {
        if (threadIdx.x < st) red[threadIdx.x] += red[threadIdx.x + st];
        __syncthreads();
    }
    const float var = red[0] / (float)D - mu * mu;
    const float inv = rsqrtf(var + 1e-5f);
#pragma unroll
    for (int j = 0; j < 4; ++j) {
        const int i = threadIdx.x + 256 * j;
        const float o = (vals[j] - mu) * inv * g[i] + bt[i];
        if (out_f32) out_f32[(size_t)m * D + i] = o;
        if (out_bf)  out_bf [(size_t)m * D + i] = f2bf(o);
    }
}

// ---------------------------------------------------------------------------
// Host-side orchestration
// ---------------------------------------------------------------------------
extern "C" void kernel_launch(void* const* d_in, const int* in_sizes, int n_in,
                              void* d_out, int out_size, void* d_ws, size_t ws_size,
                              hipStream_t stream) {
    (void)in_sizes; (void)n_in; (void)out_size; (void)ws_size;
    const int Bc = 2, Sc = 2048, Dc = 1024, Hc = 16, Fc = 4096;
    const int Mc = Bc * Sc;                          // 4096 rows

    const float* enc = (const float*)d_in[0];
    const float* dec = (const float*)d_in[1];
    const float* sa_wq = (const float*)d_in[2];  const float* sa_bq = (const float*)d_in[3];
    const float* sa_wk = (const float*)d_in[4];  const float* sa_bk = (const float*)d_in[5];
    const float* sa_wv = (const float*)d_in[6];  const float* sa_bv = (const float*)d_in[7];
    const float* sa_wo = (const float*)d_in[8];  const float* sa_bo = (const float*)d_in[9];
    const float* sa_g  = (const float*)d_in[10]; const float* sa_bt = (const float*)d_in[11];
    const float* ca_wq = (const float*)d_in[12]; const float* ca_bq = (const float*)d_in[13];
    const float* ca_wk = (const float*)d_in[14]; const float* ca_bk = (const float*)d_in[15];
    const float* ca_wv = (const float*)d_in[16]; const float* ca_bv = (const float*)d_in[17];
    const float* ca_wo = (const float*)d_in[18]; const float* ca_bo = (const float*)d_in[19];
    const float* ca_g  = (const float*)d_in[20]; const float* ca_bt = (const float*)d_in[21];
    const float* f_w1 = (const float*)d_in[22];  const float* f_b1 = (const float*)d_in[23];
    const float* f_w2 = (const float*)d_in[24];  const float* f_b2 = (const float*)d_in[25];
    const float* f_g  = (const float*)d_in[26];  const float* f_bt = (const float*)d_in[27];

    // ---- workspace layout ----
    char* wsp = (char*)d_ws;
    size_t off = 0;
    auto grab = [&](size_t bytes) -> void* {
        void* p = wsp + off;
        off += (bytes + 255) & ~(size_t)255;
        return p;
    };
    typedef unsigned short bfr;
    const size_t MD = (size_t)Mc * Dc, DD = (size_t)Dc * Dc, DF = (size_t)Dc * Fc, MF = (size_t)Mc * Fc;
    bfr* sa_wq_b = (bfr*)grab(DD * 2); bfr* sa_wk_b = (bfr*)grab(DD * 2);
    bfr* sa_wv_b = (bfr*)grab(DD * 2); bfr* sa_wo_b = (bfr*)grab(DD * 2);
    bfr* ca_wq_b = (bfr*)grab(DD * 2); bfr* ca_wk_b = (bfr*)grab(DD * 2);
    bfr* ca_wv_b = (bfr*)grab(DD * 2); bfr* ca_wo_b = (bfr*)grab(DD * 2);
    bfr* w1_b = (bfr*)grab(DF * 2);    bfr* w2_b = (bfr*)grab(DF * 2);
    bfr* enc_b = (bfr*)grab(MD * 2);   bfr* dec_b = (bfr*)grab(MD * 2);
    bfr* q_b = (bfr*)grab(MD * 2); bfr* k_b = (bfr*)grab(MD * 2); bfr* v_b = (bfr*)grab(MD * 2);
    bfr* ao_b = (bfr*)grab(MD * 2);
    bfr* x_b = (bfr*)grab(MD * 2);  bfr* y_b = (bfr*)grab(MD * 2);
    bfr* h1_b = (bfr*)grab(MF * 2);
    float* tmp_f = (float*)grab(MD * 4);
    float* x_f   = (float*)grab(MD * 4);
    float* y_f   = (float*)grab(MD * 4);

    auto cvt = [&](const float* src, bfr* dst, size_t n) {
        int blocks = (int)((n + 255) / 256);
        cvt_f32_bf16<<<blocks, 256, 0, stream>>>(src, dst, (long long)n);
    };
    auto gemm = [&](const bfr* A, const bfr* W, const float* bias,
                    bfr* obf, float* of32, int M, int N, int K, int relu) {
        dim3 grid(N / 128, M / 128);
        gemm_bf16_wmma<<<grid, 256, 0, stream>>>(A, W, bias, obf, of32, M, N, K, relu);
    };

    // ---- convert inputs & weights to bf16 ----
    cvt(enc, enc_b, MD);     cvt(dec, dec_b, MD);
    cvt(sa_wq, sa_wq_b, DD); cvt(sa_wk, sa_wk_b, DD);
    cvt(sa_wv, sa_wv_b, DD); cvt(sa_wo, sa_wo_b, DD);
    cvt(ca_wq, ca_wq_b, DD); cvt(ca_wk, ca_wk_b, DD);
    cvt(ca_wv, ca_wv_b, DD); cvt(ca_wo, ca_wo_b, DD);
    cvt(f_w1, w1_b, DF);     cvt(f_w2, w2_b, DF);

    const dim3 fgrid(Sc / 128, Bc * Hc);

    // ---- masked self-attention ----
    gemm(dec_b, sa_wq_b, sa_bq, q_b, nullptr, Mc, Dc, Dc, 0);
    gemm(dec_b, sa_wk_b, sa_bk, k_b, nullptr, Mc, Dc, Dc, 0);
    gemm(dec_b, sa_wv_b, sa_bv, v_b, nullptr, Mc, Dc, Dc, 0);
    flash_attn_wmma<<<fgrid, 256, 0, stream>>>(q_b, k_b, v_b, ao_b, Sc, Dc, Hc, 1);
    gemm(ao_b, sa_wo_b, sa_bo, nullptr, tmp_f, Mc, Dc, Dc, 0);
    add_ln_kernel<<<Mc, 256, 0, stream>>>(tmp_f, dec, sa_g, sa_bt, x_f, x_b, Dc);

    // ---- cross-attention ----
    gemm(x_b,   ca_wq_b, ca_bq, q_b, nullptr, Mc, Dc, Dc, 0);
    gemm(enc_b, ca_wk_b, ca_bk, k_b, nullptr, Mc, Dc, Dc, 0);
    gemm(enc_b, ca_wv_b, ca_bv, v_b, nullptr, Mc, Dc, Dc, 0);
    flash_attn_wmma<<<fgrid, 256, 0, stream>>>(q_b, k_b, v_b, ao_b, Sc, Dc, Hc, 0);
    gemm(ao_b, ca_wo_b, ca_bo, nullptr, tmp_f, Mc, Dc, Dc, 0);
    add_ln_kernel<<<Mc, 256, 0, stream>>>(tmp_f, x_f, ca_g, ca_bt, y_f, y_b, Dc);

    // ---- FFN ----
    gemm(y_b,  w1_b, f_b1, h1_b, nullptr, Mc, Fc, Dc, 1);   // ReLU fused
    gemm(h1_b, w2_b, f_b2, nullptr, tmp_f, Mc, Dc, Fc, 0);
    add_ln_kernel<<<Mc, 256, 0, stream>>>(tmp_f, y_f, f_g, f_bt, (float*)d_out, nullptr, Dc);
}
